// MIPNetwork_89240830476891
// MI455X (gfx1250) — matrix operand order repaired and checked by reading, not compiled, not run on hardware
//
#include <hip/hip_runtime.h>
#include <hip/hip_bf16.h>
#include <math.h>

#define FM 64
#define BITS 16
#define STEPS 3
#define PN_BLOCKS 120

typedef __attribute__((ext_vector_type(16))) __bf16 v16bf;
typedef __attribute__((ext_vector_type(8)))  __bf16 v8bf;
typedef __attribute__((ext_vector_type(8)))  float  v8f;

static __device__ __forceinline__ __bf16 f2bf(float f) {
  union { float f; unsigned u; } v; v.f = f;
  unsigned r = v.u + 0x7FFFu + ((v.u >> 16) & 1u);   // round-to-nearest-even
  return __builtin_bit_cast(__bf16, (unsigned short)(r >> 16));
}
static __device__ __forceinline__ float bf2f(__bf16 h) {
  unsigned short u = __builtin_bit_cast(unsigned short, h);
  union { unsigned u; float f; } v; v.u = ((unsigned)u) << 16;
  return v.f;
}

// ---------------------------------------------------------------- fills
__global__ void k_fill_f32(float* p, float v, long long n) {
  long long i = (long long)blockIdx.x * blockDim.x + threadIdx.x;
  if (i < n) p[i] = v;
}
__global__ void k_fill_u16(unsigned short* p, unsigned short v, long long n) {
  long long i = (long long)blockIdx.x * blockDim.x + threadIdx.x;
  if (i < n) p[i] = v;
}

// --------------------------------------------------------- f32 -> bf16
__global__ void k_cvt(const float* __restrict__ in, __bf16* __restrict__ out,
                      long long n) {
  long long i = (long long)blockIdx.x * blockDim.x + threadIdx.x;
  if (i < n) out[i] = f2bf(in[i]);
}

// -------------------------------------------- weight pack to frag order
// Wp[(((nt*Kc + kc)*32 + lane)*16 + e)] = W[(kc*32 + (lane>=16?8:0) + (e&7) + (e>=8?16:0))*N + nt*16 + (lane&15)]
__global__ void k_pack_w(const float* __restrict__ W, __bf16* __restrict__ Wp,
                         int K, int N) {
  int i = blockIdx.x * blockDim.x + threadIdx.x;
  if (i >= K * N) return;
  int e    = i & 15;
  int lane = (i >> 4) & 31;
  int rest = i >> 9;
  int Kc   = K >> 5;
  int kc   = rest % Kc;
  int nt   = rest / Kc;
  int k    = kc * 32 + ((lane >> 4) << 3) + (e & 7) + ((e >> 3) << 4);
  int col  = nt * 16 + (lane & 15);
  Wp[i] = f2bf(W[k * N + col]);
}

// ------------------------------------------------- constraints embed L1
__global__ void k_pc1(const float* __restrict__ cond, const float* __restrict__ w1,
                      const float* __restrict__ b1, __bf16* __restrict__ h, int C) {
  int i = blockIdx.x * blockDim.x + threadIdx.x;
  if (i >= C * FM) return;
  int c = i >> 6, f = i & 63;
  float x = cond[c] * w1[f] + b1[f];
  h[i] = f2bf(x > 0.f ? x : 0.f);
}

// ---------------------------------------------------------------- GEMM
// Y[M,N] = epi( concat_K(X0|X1|rank1) @ W + bias ), operands bf16, acc f32.
// One 16x16 tile per wave, K fully unrolled -> 2*NSRC v_wmma per kernel.
// EPI: 0=linear(bf16 out) 1=relu(bf16 out) 2=sigmoid+noise(f32 out)
template <int NSRC, bool RANK1, int EPI>
__global__ void k_gemm(const __bf16* __restrict__ X0, const __bf16* __restrict__ X1,
                       const float* __restrict__ r1col, const float* __restrict__ r1row,
                       const __bf16* __restrict__ Wp, const float* __restrict__ bias,
                       void* __restrict__ Yv, const float* __restrict__ noise,
                       int M, int N) {
  int lane = threadIdx.x & 31;
  int wave = threadIdx.x >> 5;
  int ntiles = N >> 4;
  int tile = blockIdx.x * 4 + wave;            // wave-uniform
  if (tile >= (M >> 4) * ntiles) return;
  int mt = tile / ntiles;
  int nt = tile % ntiles;

  int half = lane >> 4;
  int rowA = mt * 16 + (lane & 15);
  int colB = nt * 16 + (lane & 15);
  int kA   = half * 8;                          // ISA 7.12.2 A layout
  const __bf16* wbase = Wp + ((size_t)(nt * (NSRC * 2)) * 32 + lane) * 16;

  v8f acc = {};
#pragma unroll
  for (int s = 0; s < NSRC; ++s) {
#pragma unroll
    for (int c = 0; c < 64; c += 32) {
      v16bf a;
      if (RANK1 && s == NSRC - 1) {             // o2v = obj_vals[row]*objectives[k]
        float cv = r1col[rowA];
#pragma unroll
        for (int e = 0; e < 8; ++e) a[e]     = f2bf(cv * r1row[c + kA + e]);
#pragma unroll
        for (int e = 0; e < 8; ++e) a[8 + e] = f2bf(cv * r1row[c + 16 + kA + e]);
      } else {
        const __bf16* Xs = (s == 0) ? X0 : X1;
        const v8bf* p0 = (const v8bf*)(Xs + (long long)rowA * FM + c + kA);
        const v8bf* p1 = (const v8bf*)(Xs + (long long)rowA * FM + c + 16 + kA);
        v8bf lo = *p0, hi = *p1;
        a = __builtin_shufflevector(lo, hi, 0, 1, 2, 3, 4, 5, 6, 7,
                                    8, 9, 10, 11, 12, 13, 14, 15);
      }
      v16bf b = *(const v16bf*)(wbase + (size_t)(s * 2 + (c >> 5)) * 512);
      acc = __builtin_amdgcn_wmma_f32_16x16x32_bf16(
          false, a, false, b, (short)0, acc, false, false);
    }
  }

  float bcol = bias[colB];
  int rbase = mt * 16 + half * 8;
  if (EPI == 2) {
    float* Y = (float*)Yv;
#pragma unroll
    for (int p = 0; p < 8; ++p) {
      int row = rbase + p;
      float v = acc[p] + bcol + noise[(long long)row * N + colB];
      Y[(long long)row * N + colB] = 1.0f / (1.0f + __expf(-v));
    }
  } else {
    __bf16* Y = (__bf16*)Yv;
#pragma unroll
    for (int p = 0; p < 8; ++p) {
      int row = rbase + p;
      float v = acc[p] + bcol;
      if (EPI == 1) v = v > 0.f ? v : 0.f;
      Y[(long long)row * N + colB] = f2bf(v);
    }
  }
}

// ----------------------------------------------------------- PairNorm
__global__ void k_pn_partial(const __bf16* __restrict__ X, int M,
                             float* pcs, float* pss) {
  __shared__ float s_cs[256];
  __shared__ float s_sq[256];
  int tid = threadIdx.x;
  long long n = (long long)M * FM;
  float cs = 0.f, sq = 0.f;
  for (long long i = (long long)blockIdx.x * 256 + tid; i < n;
       i += (long long)gridDim.x * 256) {       // stride % 64 == 0 -> fixed feature
    float x = bf2f(X[i]);
    cs += x; sq += x * x;
  }
  s_cs[tid] = cs; s_sq[tid] = sq;
  __syncthreads();
  if (tid < 64)
    pcs[blockIdx.x * 64 + tid] =
        s_cs[tid] + s_cs[tid + 64] + s_cs[tid + 128] + s_cs[tid + 192];
  for (int st = 128; st > 0; st >>= 1) {
    if (tid < st) s_sq[tid] += s_sq[tid + st];
    __syncthreads();
  }
  if (tid == 0) pss[blockIdx.x] = s_sq[0];
}

__global__ void k_pn_finalize(const float* pcs, const float* pss, int M,
                              int nblocks, float* stats) {
  __shared__ float smu2[64];
  int tid = threadIdx.x;   // 64
  float c = 0.f;
  for (int b = 0; b < nblocks; ++b) c += pcs[b * 64 + tid];
  float mu = c / (float)M;
  stats[tid] = mu;
  smu2[tid] = mu * mu;
  __syncthreads();
  if (tid == 0) {
    float ss = 0.f;
    for (int b = 0; b < nblocks; ++b) ss += pss[b];
    float m2 = 0.f;
    for (int f = 0; f < 64; ++f) m2 += smu2[f];
    stats[64] = rsqrtf(1e-6f + (ss - (float)M * m2) / (float)M);
  }
}

__global__ void k_pn_apply(__bf16* X, int M, const float* __restrict__ stats) {
  long long i = (long long)blockIdx.x * 256 + threadIdx.x;
  if (i >= (long long)M * FM) return;
  X[i] = f2bf((bf2f(X[i]) - stats[i & 63]) * stats[64]);
}

// ---------------------------------------------------- SpMM scatter-add
__global__ void k_scatter(const int* __restrict__ srcIdx, const int* __restrict__ dstIdx,
                          const float* __restrict__ vals, const __bf16* __restrict__ Xsrc,
                          float* Ydst, long long nnz) {
  long long i = (long long)blockIdx.x * 256 + threadIdx.x;
  if (i >= nnz * FM) return;
  long long e = i >> 6;
  int f = (int)(i & 63);
  atomicAdd(&Ydst[(long long)dstIdx[e] * FM + f],
            vals[e] * bf2f(Xsrc[(long long)srcIdx[e] * FM + f]));
}

// --------------------------------------------------------------- v2o
__global__ void k_v2o(const int* __restrict__ obj_rows, const float* __restrict__ obj_vals,
                      const __bf16* __restrict__ variables, float* v2o, int V) {
  __shared__ float s[256];
  int tid = threadIdx.x;
  long long n = (long long)V * FM;
  float acc = 0.f;
  for (long long i = (long long)blockIdx.x * 256 + tid; i < n;
       i += (long long)gridDim.x * 256) {
    long long v = i >> 6; int f = (int)(i & 63);
    acc += obj_vals[v] * bf2f(variables[(long long)obj_rows[v] * FM + f]);
  }
  s[tid] = acc;
  __syncthreads();
  if (tid < 64)
    atomicAdd(&v2o[tid], s[tid] + s[tid + 64] + s[tid + 128] + s[tid + 192]);
}

// ------------------------------------------- objectives MLP + PairNorm
__global__ void k_obj_update(float* obj, const float* __restrict__ v2o,
                             const float* __restrict__ w1, const float* __restrict__ b1,
                             const float* __restrict__ w2, const float* __restrict__ b2) {
  __shared__ float cat[128];
  __shared__ float h[64];
  __shared__ float red[64];
  int tid = threadIdx.x;   // 64
  cat[tid] = obj[tid];
  cat[64 + tid] = v2o[tid];
  __syncthreads();
  float a = b1[tid];
  for (int k = 0; k < 128; ++k) a += cat[k] * w1[k * 64 + tid];
  h[tid] = a > 0.f ? a : 0.f;
  __syncthreads();
  float y = b2[tid];
  for (int k = 0; k < 64; ++k) y += h[k] * w2[k * 64 + tid];
  float cent = y - y;                 // 1-row PairNorm: centered == 0 exactly
  red[tid] = cent * cent;
  __syncthreads();
  for (int st = 32; st > 0; st >>= 1) {
    if (tid < st) red[tid] += red[tid + st];
    __syncthreads();
  }
  obj[tid] = cent * rsqrtf(1e-6f + red[0]);
}

// --------------------------------------------------------------- decs
__global__ void k_decs(const float* __restrict__ bins, float* decs, int V) {
  int v = blockIdx.x * blockDim.x + threadIdx.x;
  if (v >= V) return;
  const float* r = bins + (long long)v * BITS;
  float s = 0.f, p = 1.f;
  for (int b = 0; b < BITS; ++b) { s += p * r[b]; p *= 2.f; }
  decs[v] = s;
}

// ====================================================================
extern "C" void kernel_launch(void* const* d_in, const int* in_sizes, int n_in,
                              void* d_out, int out_size, void* d_ws, size_t ws_size,
                              hipStream_t stream) {
  const int*   adj_rows = (const int*)d_in[0];
  const int*   adj_cols = (const int*)d_in[1];
  const float* adj_vals = (const float*)d_in[2];
  const int*   obj_rows = (const int*)d_in[3];
  const float* obj_vals = (const float*)d_in[4];
  const float* cond     = (const float*)d_in[5];
  const float* noise    = (const float*)d_in[6];
  const float* pc_w1 = (const float*)d_in[7];
  const float* pc_b1 = (const float*)d_in[8];
  const float* pc_w2 = (const float*)d_in[9];
  const float* pc_b2 = (const float*)d_in[10];
  const float* cu_w1 = (const float*)d_in[11];
  const float* cu_b1 = (const float*)d_in[12];
  const float* cu_w2 = (const float*)d_in[13];
  const float* cu_b2 = (const float*)d_in[14];
  const float* vu_w1 = (const float*)d_in[15];
  const float* vu_b1 = (const float*)d_in[16];
  const float* vu_w2 = (const float*)d_in[17];
  const float* vu_b2 = (const float*)d_in[18];
  const float* ou_w1 = (const float*)d_in[19];
  const float* ou_b1 = (const float*)d_in[20];
  const float* ou_w2 = (const float*)d_in[21];
  const float* ou_b2 = (const float*)d_in[22];
  const float* out_w1 = (const float*)d_in[23];
  const float* out_b1 = (const float*)d_in[24];
  const float* out_w2 = (const float*)d_in[25];
  const float* out_b2 = (const float*)d_in[26];

  const long long NNZ = in_sizes[0];
  const int V = in_sizes[3];
  const int C = in_sizes[5];

  // -------- workspace carve (256B aligned slots)
  char* base = (char*)d_ws;
  size_t off = 0;
  auto carve = [&](size_t bytes) -> char* {
    char* p = base + off;
    off += (bytes + 255) & ~(size_t)255;
    return p;
  };
  __bf16* variables = (__bf16*)carve((size_t)V * FM * 2);
  __bf16* constraints = (__bf16*)carve((size_t)C * FM * 2);
  __bf16* Hb       = (__bf16*)carve((size_t)V * FM * 2);
  __bf16* v2c_bf   = (__bf16*)carve((size_t)C * FM * 2);
  __bf16* c2v_bf   = (__bf16*)carve((size_t)V * FM * 2);
  float*  v2c_f    = (float*)carve((size_t)C * FM * 4);
  float*  c2v_f    = (float*)carve((size_t)V * FM * 4);
  __bf16* pc_w2p  = (__bf16*)carve(64 * 64 * 2);
  __bf16* cu_w1p  = (__bf16*)carve(128 * 64 * 2);
  __bf16* cu_w2p  = (__bf16*)carve(64 * 64 * 2);
  __bf16* vu_w1p  = (__bf16*)carve(192 * 64 * 2);
  __bf16* vu_w2p  = (__bf16*)carve(64 * 64 * 2);
  __bf16* out_w1p = (__bf16*)carve(64 * 64 * 2);
  __bf16* out_w2p = (__bf16*)carve(64 * 16 * 2);
  float* objv  = (float*)carve(64 * 4);
  float* v2o   = (float*)carve(64 * 4);
  float* pcs   = (float*)carve(PN_BLOCKS * 64 * 4);
  float* pss   = (float*)carve(PN_BLOCKS * 4);
  float* stats = (float*)carve(65 * 4);

  float* out_bins = (float*)d_out;                 // [3][V][16]
  float* out_decs = out_bins + 3LL * V * BITS;     // [3][V]

  auto blocks = [](long long n) { return (int)((n + 255) / 256); };
  auto pairnorm = [&](__bf16* X, int M) {
    k_pn_partial<<<PN_BLOCKS, 256, 0, stream>>>(X, M, pcs, pss);
    k_pn_finalize<<<1, 64, 0, stream>>>(pcs, pss, M, PN_BLOCKS, stats);
    k_pn_apply<<<blocks((long long)M * FM), 256, 0, stream>>>(X, M, stats);
  };
  auto gblk = [](int M, int N) { return ((M / 16) * (N / 16) + 3) / 4; };

  // -------- pack weights (once per launch; tiny)
  k_pack_w<<<blocks(64 * 64), 256, 0, stream>>>(pc_w2, pc_w2p, 64, 64);
  k_pack_w<<<blocks(128 * 64), 256, 0, stream>>>(cu_w1, cu_w1p, 128, 64);
  k_pack_w<<<blocks(64 * 64), 256, 0, stream>>>(cu_w2, cu_w2p, 64, 64);
  k_pack_w<<<blocks(192 * 64), 256, 0, stream>>>(vu_w1, vu_w1p, 192, 64);
  k_pack_w<<<blocks(64 * 64), 256, 0, stream>>>(vu_w2, vu_w2p, 64, 64);
  k_pack_w<<<blocks(64 * 64), 256, 0, stream>>>(out_w1, out_w1p, 64, 64);
  k_pack_w<<<blocks(64 * 16), 256, 0, stream>>>(out_w2, out_w2p, 64, 16);

  // -------- init
  k_fill_u16<<<blocks((long long)V * FM), 256, 0, stream>>>(
      (unsigned short*)variables, (unsigned short)0x3F80, (long long)V * FM);  // 1.0bf
  k_fill_f32<<<1, 64, 0, stream>>>(objv, 1.0f, 64);
  k_pc1<<<blocks((long long)C * FM), 256, 0, stream>>>(cond, pc_w1, pc_b1, Hb, C);
  k_gemm<1, false, 0><<<gblk(C, FM), 128, 0, stream>>>(
      Hb, nullptr, nullptr, nullptr, pc_w2p, pc_b2, constraints, nullptr, C, FM);
  pairnorm(constraints, C);

  const long long tn = NNZ * FM;
  for (int i = 0; i < STEPS; ++i) {
    // objectives branch
    k_fill_f32<<<1, 64, 0, stream>>>(v2o, 0.f, 64);
    k_v2o<<<256, 256, 0, stream>>>(obj_rows, obj_vals, variables, v2o, V);
    k_obj_update<<<1, 64, 0, stream>>>(objv, v2o, ou_w1, ou_b1, ou_w2, ou_b2);
    // v2c = adj^T @ variables
    k_fill_f32<<<blocks((long long)C * FM), 256, 0, stream>>>(v2c_f, 0.f, (long long)C * FM);
    k_scatter<<<blocks(tn), 256, 0, stream>>>(adj_rows, adj_cols, adj_vals,
                                              variables, v2c_f, NNZ);
    k_cvt<<<blocks((long long)C * FM), 256, 0, stream>>>(v2c_f, v2c_bf, (long long)C * FM);
    // constraints = pair_norm(mlp([constraints, v2c]))
    k_gemm<2, false, 1><<<gblk(C, FM), 128, 0, stream>>>(
        constraints, v2c_bf, nullptr, nullptr, cu_w1p, cu_b1, Hb, nullptr, C, FM);
    k_gemm<1, false, 0><<<gblk(C, FM), 128, 0, stream>>>(
        Hb, nullptr, nullptr, nullptr, cu_w2p, cu_b2, constraints, nullptr, C, FM);
    pairnorm(constraints, C);
    // c2v = adj @ constraints
    k_fill_f32<<<blocks((long long)V * FM), 256, 0, stream>>>(c2v_f, 0.f, (long long)V * FM);
    k_scatter<<<blocks(tn), 256, 0, stream>>>(adj_cols, adj_rows, adj_vals,
                                              constraints, c2v_f, NNZ);
    k_cvt<<<blocks((long long)V * FM), 256, 0, stream>>>(c2v_f, c2v_bf, (long long)V * FM);
    // variables = pair_norm(mlp([variables, c2v, o2v]))  (o2v rank-1 fused)
    k_gemm<3, true, 1><<<gblk(V, FM), 128, 0, stream>>>(
        variables, c2v_bf, obj_vals, objv, vu_w1p, vu_b1, Hb, nullptr, V, FM);
    k_gemm<1, false, 0><<<gblk(V, FM), 128, 0, stream>>>(
        Hb, nullptr, nullptr, nullptr, vu_w2p, vu_b2, variables, nullptr, V, FM);
    pairnorm(variables, V);
    // bins = sigmoid(mlp(variables) + noise[i]); decs = sum(2^b * bins)
    k_gemm<1, false, 1><<<gblk(V, FM), 128, 0, stream>>>(
        variables, nullptr, nullptr, nullptr, out_w1p, out_b1, Hb, nullptr, V, FM);
    float* bins_i = out_bins + (long long)i * V * BITS;
    k_gemm<1, false, 2><<<gblk(V, BITS), 128, 0, stream>>>(
        Hb, nullptr, nullptr, nullptr, out_w2p, out_b2, bins_i,
        noise + (long long)i * V * BITS, V, BITS);
    k_decs<<<blocks(V), 256, 0, stream>>>(bins_i, out_decs + (long long)i * V, V);
  }
}
